// SemanticConnectivityLoss_12592844112075
// MI455X (gfx1250) — matrix operand order repaired and checked by reading, not compiled, not run on hardware
//
#include <hip/hip_runtime.h>
#include <cstdint>

typedef __attribute__((ext_vector_type(16))) _Float16     v16h;
typedef __attribute__((ext_vector_type(8)))  float        v8f;
typedef __attribute__((ext_vector_type(8)))  unsigned int v8u;

#define N_IMG   4
#define HW      (1024 * 1024)
#define P_CLS   10
#define L_CLS   6
#define BPI     256                 // blocks per image
#define THREADS 256
#define WAVES   8                   // wave32 -> 8 waves/block
#define PPB     (HW / BPI)          // 4096 pixels per block
#define PPW     (PPB / WAVES)       // 512 pixels per wave
#define CHUNKS  (PPW / 32)          // 16 WMMA k-chunks per wave

__device__ __forceinline__ unsigned int bperm(int srcLane, unsigned int v) {
  return (unsigned int)__builtin_amdgcn_ds_bpermute(srcLane << 2, (int)v);
}

union AB16 { v8u u; v16h h; };

// Stage 1: one-hot GEMM via v_wmma_f32_16x16x32_f16.
//   C[l][p]  (l<6, p<10) = sum pred * (lconn==l) * (pconn==p)
//   C[l][10]             = count(lconn==l)   (B column 10 is all-ones)
__global__ __launch_bounds__(THREADS) void sc_partial_kernel(
    const float* __restrict__ pred,
    const int*   __restrict__ pconn,
    const int*   __restrict__ lconn,
    float*       __restrict__ part)
{
  __shared__ float wsum[WAVES][96];

  const int lane = threadIdx.x & 31;
  const int wv   = threadIdx.x >> 5;
  const int img  = blockIdx.x / BPI;
  const int blk  = blockIdx.x % BPI;
  const int half = lane >> 4;
  const unsigned int m = (unsigned int)(lane & 15);       // A row this lane holds
  const unsigned int n = (unsigned int)(lane & 15);       // B col this lane holds
  const unsigned int base10 = (n == 10u) ? 0x3C003C00u : 0u; // col10 = 1.0h,1.0h

  const long base = (long)img * HW + (long)blk * PPB + (long)wv * PPW;

  v8f acc = {};
  for (int c = 0; c < CHUNKS; ++c) {
    const long idx = base + (long)c * 32 + lane;
    const float pv = pred[idx];
    const unsigned int lc = (unsigned int)lconn[idx];
    const unsigned int pc = (unsigned int)pconn[idx];
    const unsigned short hb = __builtin_bit_cast(unsigned short, (_Float16)pv);
    const unsigned int packed = (lc & 0xFFu) | ((pc & 0xFFu) << 8) |
                                ((unsigned int)hb << 16);

    AB16 A, B;
    #pragma unroll
    for (int v = 0; v < 8; ++v) {
      // A k order used here (contiguous, same as B): lanes0-15 hold K=0..15, lanes16-31 hold K=16..31
      const int kA = 16 * half + 2 * v;
      // B k order used here (contiguous, same as A): lanes0-15 hold K=0..15, lanes16-31 hold K=16..31
      const int kB = 16 * half + 2 * v;

      const unsigned int pA0 = bperm(kA,     packed);
      const unsigned int pA1 = bperm(kA + 1, packed);
      const unsigned int pB0 = bperm(kB,     packed);
      const unsigned int pB1 = bperm(kB + 1, packed);

      const unsigned int a_lo = ((pA0 & 0xFFu) == m) ? 0x00003C00u : 0u;
      const unsigned int a_hi = ((pA1 & 0xFFu) == m) ? 0x3C000000u : 0u;
      A.u[v] = a_lo | a_hi;

      const unsigned int b_lo = (((pB0 >> 8) & 0xFFu) == n) ? (pB0 >> 16)         : 0u;
      const unsigned int b_hi = (((pB1 >> 8) & 0xFFu) == n) ? (pB1 & 0xFFFF0000u) : 0u;
      B.u[v] = base10 | b_lo | b_hi;
    }
    // D = A x B + C   (f32 accumulate, loop-carried)
    acc = __builtin_amdgcn_wmma_f32_16x16x32_f16(
        /*neg_a=*/false, A.h, /*neg_b=*/false, B.h,
        /*c_mod=*/(short)0, acc, /*reuse_a=*/false, /*reuse_b=*/false);
  }

  // C layout: lanes0-15: VGPR j -> row M=j; lanes16-31: M=j+8; col N=lane&15.
  // Only rows 0..5 are meaningful (rows >=6 of A are zero).
  #pragma unroll
  for (int j = 0; j < 8; ++j) {
    const int M = j + 8 * half;
    if (M < L_CLS) wsum[wv][M * 16 + (int)n] = acc[j];
  }
  __syncthreads();

  // Deterministic per-block reduction across the 8 waves (fixed order, no atomics).
  for (int t = threadIdx.x; t < 96; t += THREADS) {
    float s = 0.f;
    #pragma unroll
    for (int w = 0; w < WAVES; ++w) s += wsum[w][t];
    part[(long)blockIdx.x * 96 + t] = s;
  }
}

// Stage 2: fixed-order partial reduction + exact reference scalar math.
__global__ __launch_bounds__(THREADS) void sc_finalize_kernel(
    const float* __restrict__ part, float* __restrict__ out)
{
  __shared__ float fin[N_IMG * 96];
  __shared__ float lossv[N_IMG];

  for (int e = threadIdx.x; e < N_IMG * 96; e += THREADS) {
    const int img = e / 96, ent = e % 96;
    float s = 0.f;
    for (int b = 0; b < BPI; ++b)
      s += part[((long)img * BPI + b) * 96 + ent];
    fin[e] = s;
  }
  __syncthreads();

  if (threadIdx.x < N_IMG) {
    const float* F = &fin[threadIdx.x * 96];   // F[l*16 + p], p==10 -> label count
    float la[L_CLS], pa[P_CLS];
    for (int l = 0; l < L_CLS; ++l) la[l] = F[l * 16 + 10];
    for (int p = 0; p < P_CLS; ++p) {
      float s = 0.f;
      for (int l = 0; l < L_CLS; ++l) s += F[l * 16 + p];
      pa[p] = s;                               // pred_area
    }
    float pair_sum = 0.f, colsum[P_CLS];
    for (int p = 0; p < P_CLS; ++p) colsum[p] = 0.f;
    for (int l = 1; l < L_CLS; ++l) {
      float pcn = 0.f, pnum = 0.f;
      for (int p = 1; p < P_CLS; ++p) {
        const float iv = F[l * 16 + p];
        const float un = la[l] + pa[p] - iv;
        const float iou = (iv == 0.f) ? 0.f : ((un == 0.f) ? 1.f : iv / un);
        pcn += iou;
        if (iou != 0.f) pnum += 1.f;
        colsum[p] += iou;
      }
      pair_sum += (pnum > 0.f) ? (pcn / pnum) : 0.f;
    }
    float lone = 0.f;
    for (int p = 1; p < P_CLS; ++p) if (colsum[p] == 0.f) lone += 1.f;
    lossv[threadIdx.x] = 1.f - pair_sum / ((float)(L_CLS - 1) + lone);
  }
  __syncthreads();

  if (threadIdx.x == 0) {
    float s = 0.f;
    for (int i = 0; i < N_IMG; ++i) s += lossv[i];
    out[0] = s / (float)N_IMG;
  }
}

extern "C" void kernel_launch(void* const* d_in, const int* in_sizes, int n_in,
                              void* d_out, int out_size, void* d_ws, size_t ws_size,
                              hipStream_t stream) {
  const float* pred  = (const float*)d_in[0];
  const int*   pconn = (const int*)d_in[1];
  const int*   lconn = (const int*)d_in[2];
  float* part = (float*)d_ws;                 // 1024 blocks * 96 floats = 384 KB

  sc_partial_kernel<<<N_IMG * BPI, THREADS, 0, stream>>>(pred, pconn, lconn, part);
  sc_finalize_kernel<<<1, THREADS, 0, stream>>>(part, (float*)d_out);
}
